// EncoderLayer_21388937134140
// MI455X (gfx1250) — matrix operand order repaired and checked by reading, not compiled, not run on hardware
//
#include <hip/hip_runtime.h>

// Problem constants (match reference)
#define BATCH 8
#define NTOK  1024
#define DM    512
#define NH    8
#define DK    64
#define DFF   2048
#define ROWS  (BATCH * NTOK)   // 8192

typedef __attribute__((ext_vector_type(16))) _Float16 v16h;
typedef __attribute__((ext_vector_type(8)))  float    v8f;

union Frag { v16h v; unsigned int u[8]; };

__device__ __forceinline__ v8f wmma_f16(const Frag& a, const Frag& b, v8f c) {
  // 8 args: (neg_a, A, neg_b, B, c_mod, C, reuse_a, reuse_b)
  return __builtin_amdgcn_wmma_f32_16x16x32_f16(false, a.v, false, b.v,
                                                (short)0, c, false, false);
}

// gfx1250 async global->LDS copy (ASYNCcnt-tracked), GV addressing mode.
// ldsaddr = low 32 bits of a generic pointer to __shared__ (== LDS byte addr).
__device__ __forceinline__ void async_copy_b128(const void* gsrc, unsigned ldsaddr) {
  asm volatile("global_load_async_to_lds_b128 %0, %1, off"
               :: "v"(ldsaddr), "v"(gsrc)
               : "memory");
}
__device__ __forceinline__ void wait_async0() {
  asm volatile("s_wait_asynccnt 0x0" ::: "memory");
}

// ---------------------------------------------------------------------------
// Elementwise: out_f16 = (b ? a + b : a)
// ---------------------------------------------------------------------------
__global__ __launch_bounds__(256) void k_addcvt(const float* __restrict__ a,
                                                const float* __restrict__ b,
                                                _Float16* __restrict__ o, int n) {
  int i = blockIdx.x * 256 + threadIdx.x;
  if (i < n) o[i] = (_Float16)(b ? (a[i] + b[i]) : a[i]);
}

// ---------------------------------------------------------------------------
// Weight transpose + f16 convert:  w[K][N] (f32) -> wt[N][K] (f16)
// ---------------------------------------------------------------------------
__global__ __launch_bounds__(256) void k_transcvt(const float* __restrict__ w,
                                                  _Float16* __restrict__ wt,
                                                  int K, int N) {
  int i = blockIdx.x * 256 + threadIdx.x;
  if (i < K * N) {
    int k = i / N, n = i - k * N;
    wt[(size_t)n * K + k] = (_Float16)w[i];
  }
}

// ---------------------------------------------------------------------------
// Tiled GEMM: C[M,N] = A[M,K] @ B[K,N] + bias, A f16 row-major, Bt = B^T f16
// row-major ([N][K]). 4 waves / block, 64x64 tile, k-step 32.
// Ping-pong LDS; next tile staged via global_load_async_to_lds_b128 while the
// current tile feeds the WMMAs. One s_wait_asynccnt + one barrier per step.
// ---------------------------------------------------------------------------
#define GM_F32 0   // f32 row-major [M,N]
#define GM_F16 1   // f16 row-major [M,N]
#define GM_QK  2   // f16 head layout [b,h,tok,d]   (row=b*1024+tok, col=h*64+d)
#define GM_VT  3   // f16 head-transposed [b,h,d,tok]

__global__ __launch_bounds__(128) void k_gemm(const _Float16* __restrict__ A,
                                              const _Float16* __restrict__ Bt,
                                              const float* __restrict__ bias,
                                              void* __restrict__ out,
                                              int M, int N, int K,
                                              int mode, int relu) {
  __shared__ unsigned int lA[2][64 * 16];
  __shared__ unsigned int lB[2][64 * 16];
  const int tid  = threadIdx.x;
  const int lane = tid & 31;
  const int wv   = tid >> 5;        // wave 0..3
  const int hf   = lane >> 4;       // lane half
  const int l15  = lane & 15;
  const int bm   = blockIdx.y * 64;
  const int bn   = blockIdx.x * 64;
  const int wm   = (wv >> 1) * 32;  // wave sub-tile
  const int wn   = (wv & 1) * 32;

  // staging assignment: thread -> (row, 16-byte chunk pair)
  const int lrow = tid >> 1;              // 0..63
  const int ldw  = (tid & 1) * 8;         // dword offset 0 or 8 within row
  const _Float16* gArow = A  + (size_t)(bm + lrow) * K + ldw * 2;
  const _Float16* gBrow = Bt + (size_t)(bn + lrow) * K + ldw * 2;
  unsigned laA[2], laB[2];
  laA[0] = (unsigned)(size_t)(void*)&lA[0][lrow * 16 + ldw];
  laA[1] = (unsigned)(size_t)(void*)&lA[1][lrow * 16 + ldw];
  laB[0] = (unsigned)(size_t)(void*)&lB[0][lrow * 16 + ldw];
  laB[1] = (unsigned)(size_t)(void*)&lB[1][lrow * 16 + ldw];

  auto stage = [&](int buf, int kb) {
    async_copy_b128(gArow + kb,     laA[buf]);
    async_copy_b128(gArow + kb + 8, laA[buf] + 16);
    async_copy_b128(gBrow + kb,     laB[buf]);
    async_copy_b128(gBrow + kb + 8, laB[buf] + 16);
  };

  const int nkb = K >> 5;
  stage(0, 0);
  wait_async0();
  __syncthreads();

  v8f acc[2][2] = {};

  for (int ib = 0; ib < nkb; ++ib) {
    const int cur = ib & 1;
    if (ib + 1 < nkb) {
      stage(cur ^ 1, (ib + 1) * 32);             // overlaps with WMMAs below
      if (ib + 2 < nkb) {                        // gfx1250 global_prefetch_b8
        __builtin_prefetch(gArow + (ib + 2) * 32, 0, 1);
        __builtin_prefetch(gBrow + (ib + 2) * 32, 0, 1);
      }
    }

    Frag af[2], bf[2];
#pragma unroll
    for (int mt = 0; mt < 2; ++mt) {             // A frag: row per lane
      const int m = wm + mt * 16 + l15;
#pragma unroll
      for (int f = 0; f < 8; ++f)
        af[mt].u[f] = lA[cur][m * 16 + hf * 8 + f];
    }
#pragma unroll
    for (int nt = 0; nt < 2; ++nt) {             // B frag: column per lane
      const int n = wn + nt * 16 + l15;
#pragma unroll
      for (int f = 0; f < 8; ++f)
        bf[nt].u[f] = lB[cur][n * 16 + hf * 8 + f];
    }
#pragma unroll
    for (int mt = 0; mt < 2; ++mt)
#pragma unroll
      for (int nt = 0; nt < 2; ++nt)
        acc[mt][nt] = wmma_f16(af[mt], bf[nt], acc[mt][nt]);

    wait_async0();     // next tile fully in LDS
    __syncthreads();   // all waves done reading lds[cur] / see lds[cur^1]
  }

#pragma unroll
  for (int mt = 0; mt < 2; ++mt) {
#pragma unroll
    for (int nt = 0; nt < 2; ++nt) {
      const int col = bn + wn + nt * 16 + l15;
      const float bb = bias ? bias[col] : 0.0f;
#pragma unroll
      for (int i = 0; i < 8; ++i) {
        const int row = bm + wm + mt * 16 + i + 8 * hf;
        float v = acc[mt][nt][i] + bb;
        if (relu) v = v > 0.0f ? v : 0.0f;
        if (mode == GM_F32) {
          ((float*)out)[(size_t)row * N + col] = v;
        } else if (mode == GM_F16) {
          ((_Float16*)out)[(size_t)row * N + col] = (_Float16)v;
        } else {
          const int b = row >> 10, t = row & 1023;
          const int h = col >> 6,  d = col & 63;
          if (mode == GM_QK)
            ((_Float16*)out)[(((size_t)(b * NH + h) * NTOK) + t) * DK + d] =
                (_Float16)v;
          else
            ((_Float16*)out)[(((size_t)(b * NH + h) * DK) + d) * NTOK + t] =
                (_Float16)v;
        }
      }
    }
  }
}

// ---------------------------------------------------------------------------
// Fused attention (online softmax), one wave per (b,h, 16-query tile).
// S = QK^T/8 + log(clip(relw)); P = softmax(S); O = P @ V.
// Q,K: [b,h,tok,d] f16.  V: [b,h,d,tok] f16 (transposed -> contiguous B frags).
// ---------------------------------------------------------------------------
__global__ __launch_bounds__(32) void k_attn(const _Float16* __restrict__ Qh,
                                             const _Float16* __restrict__ Kh,
                                             const _Float16* __restrict__ Vth,
                                             const float* __restrict__ relw,
                                             _Float16* __restrict__ outh) {
  __shared__ unsigned int lP[16 * 16];   // 16 x 32 f16 P tile
  const int bid  = blockIdx.x;
  const int qt   = bid & 63;             // query tile
  const int bh   = bid >> 6;             // b*NH + h
  const int lane = threadIdx.x;
  const int hf   = lane >> 4;
  const int l15  = lane & 15;
  const _Float16* Q  = Qh  + (size_t)bh * NTOK * DK;
  const _Float16* Kp = Kh  + (size_t)bh * NTOK * DK;
  const _Float16* Vt = Vth + (size_t)bh * DK * NTOK;
  const float*    W  = relw + (size_t)bh * NTOK * NTOK;
  const int q0 = qt * 16;

  Frag aq[2];                                          // Q 16x64 = 2 A-frags
  {
    const unsigned int* qrow = (const unsigned int*)(Q + (size_t)(q0 + l15) * DK);
#pragma unroll
    for (int ks = 0; ks < 2; ++ks)
#pragma unroll
      for (int f = 0; f < 8; ++f)
        aq[ks].u[f] = qrow[ks * 16 + hf * 8 + f];
  }

  v8f o[4] = {};
  float mrow[8], lrow[8];
#pragma unroll
  for (int i = 0; i < 8; ++i) { mrow[i] = -3.0e38f; lrow[i] = 0.0f; }

  for (int kt = 0; kt < NTOK / 32; ++kt) {     // 32 keys per step
    v8f s[2] = {};
#pragma unroll
    for (int nt = 0; nt < 2; ++nt) {
      const int key = kt * 32 + nt * 16 + l15;
      const unsigned int* krow = (const unsigned int*)(Kp + (size_t)key * DK);
#pragma unroll
      for (int ks = 0; ks < 2; ++ks) {
        Frag bkf;                              // B frag: column(=key) per lane
#pragma unroll
        for (int f = 0; f < 8; ++f) bkf.u[f] = krow[ks * 16 + hf * 8 + f];
        s[nt] = wmma_f16(aq[ks], bkf, s[nt]);
      }
    }
    // bias + online softmax (row i lives in vgpr i of half hf — C layout)
    float p0[8], p1[8], alpha[8];
#pragma unroll
    for (int i = 0; i < 8; ++i) {
      const int row = q0 + i + 8 * hf;
      const float* wr = W + (size_t)row * NTOK + kt * 32;
      float v0 = s[0][i] * 0.125f + __logf(fmaxf(wr[l15], 1e-6f));
      float v1 = s[1][i] * 0.125f + __logf(fmaxf(wr[16 + l15], 1e-6f));
      float mx = fmaxf(v0, v1);
#pragma unroll
      for (int d = 1; d < 16; d <<= 1) mx = fmaxf(mx, __shfl_xor(mx, d, 32));
      const float mn = fmaxf(mrow[i], mx);
      alpha[i] = __expf(mrow[i] - mn);
      v0 = __expf(v0 - mn);
      v1 = __expf(v1 - mn);
      float sm = v0 + v1;
#pragma unroll
      for (int d = 1; d < 16; d <<= 1) sm += __shfl_xor(sm, d, 32);
      lrow[i] = lrow[i] * alpha[i] + sm;
      mrow[i] = mn;
      p0[i] = v0; p1[i] = v1;
    }
#pragma unroll
    for (int t = 0; t < 4; ++t)
#pragma unroll
      for (int i = 0; i < 8; ++i) o[t][i] *= alpha[i];

    // C-layout -> A-layout re-stripe of P through LDS
    __syncthreads();
    _Float16* lph = (_Float16*)lP;
#pragma unroll
    for (int i = 0; i < 8; ++i) {
      lph[(i + 8 * hf) * 32 + l15]      = (_Float16)p0[i];
      lph[(i + 8 * hf) * 32 + 16 + l15] = (_Float16)p1[i];
    }
    __syncthreads();
    Frag pf;
#pragma unroll
    for (int f = 0; f < 8; ++f)
      pf.u[f] = lP[l15 * 16 + hf * 8 + f];

#pragma unroll
    for (int t = 0; t < 4; ++t) {               // O(16x64) += P(16x32) @ V(32x16)
      const int dcol = t * 16 + l15;
      const unsigned int* vrow =
          (const unsigned int*)(Vt + (size_t)dcol * NTOK) + kt * 16;
      Frag bvf;
#pragma unroll
      for (int f = 0; f < 8; ++f) bvf.u[f] = vrow[hf * 8 + f];
      o[t] = wmma_f16(pf, bvf, o[t]);
    }
  }

  const int bb = bh >> 3, hh = bh & 7;
#pragma unroll
  for (int i = 0; i < 8; ++i) {
    const float inv = 1.0f / lrow[i];
    const int row = q0 + i + 8 * hf;
#pragma unroll
    for (int t = 0; t < 4; ++t)
      outh[((size_t)(bb * NTOK + row)) * DM + hh * DK + t * 16 + l15] =
          (_Float16)(o[t][i] * inv);
  }
}

// ---------------------------------------------------------------------------
// LayerNorm with fused residual: y = LN(x + r) * g + b. One block per row.
// Writes f32 and (optionally) an f16 copy for the next GEMM.
// ---------------------------------------------------------------------------
__global__ __launch_bounds__(256) void k_ln(const float* __restrict__ x,
                                            const float* __restrict__ r,
                                            const float* __restrict__ g,
                                            const float* __restrict__ be,
                                            float* __restrict__ outf,
                                            _Float16* __restrict__ outh) {
  const int row = blockIdx.x;
  const int tid = threadIdx.x;
  const float* xr = x + (size_t)row * DM;
  const float* rr = r + (size_t)row * DM;
  float v0 = xr[tid] + rr[tid];
  float v1 = xr[tid + 256] + rr[tid + 256];
  float s = v0 + v1;
  float q = v0 * v0 + v1 * v1;
#pragma unroll
  for (int d = 1; d < 32; d <<= 1) {
    s += __shfl_xor(s, d, 32);
    q += __shfl_xor(q, d, 32);
  }
  __shared__ float ss[8], sq[8];
  const int w = tid >> 5;
  if ((tid & 31) == 0) { ss[w] = s; sq[w] = q; }
  __syncthreads();
  float S = 0.0f, Q = 0.0f;
#pragma unroll
  for (int i = 0; i < 8; ++i) { S += ss[i]; Q += sq[i]; }
  const float mu  = S * (1.0f / DM);
  const float var = Q * (1.0f / DM) - mu * mu;
  const float inv = rsqrtf(var + 1e-5f);
  const float y0 = (v0 - mu) * inv * g[tid] + be[tid];
  const float y1 = (v1 - mu) * inv * g[tid + 256] + be[tid + 256];
  outf[(size_t)row * DM + tid]       = y0;
  outf[(size_t)row * DM + tid + 256] = y1;
  if (outh) {
    outh[(size_t)row * DM + tid]       = (_Float16)y0;
    outh[(size_t)row * DM + tid + 256] = (_Float16)y1;
  }
}

// ---------------------------------------------------------------------------
extern "C" void kernel_launch(void* const* d_in, const int* in_sizes, int n_in,
                              void* d_out, int out_size, void* d_ws, size_t ws_size,
                              hipStream_t stream) {
  (void)in_sizes; (void)n_in; (void)out_size; (void)ws_size;
  const float* queries = (const float*)d_in[0];
  const float* keys    = (const float*)d_in[1];
  const float* values  = (const float*)d_in[2];
  const float* relw    = (const float*)d_in[3];
  const float* pos     = (const float*)d_in[4];
  const float* Wq = (const float*)d_in[5];  const float* bq  = (const float*)d_in[6];
  const float* Wk = (const float*)d_in[7];  const float* bk  = (const float*)d_in[8];
  const float* Wv = (const float*)d_in[9];  const float* bv  = (const float*)d_in[10];
  const float* Wo = (const float*)d_in[11]; const float* bo  = (const float*)d_in[12];
  const float* g1 = (const float*)d_in[13]; const float* be1 = (const float*)d_in[14];
  const float* W1 = (const float*)d_in[15]; const float* b1  = (const float*)d_in[16];
  const float* W2 = (const float*)d_in[17]; const float* b2  = (const float*)d_in[18];
  const float* g2 = (const float*)d_in[19]; const float* be2 = (const float*)d_in[20];

  char* ws = (char*)d_ws;
  const size_t MB = 1ull << 20;
  // lifetime-aliased workspace layout (~118 MB total)
  _Float16* qpos    = (_Float16*)(ws + 0);        // 8 MB (dead after Q gemm)
  _Float16* kpos    = (_Float16*)(ws + 8  * MB);  // 8 MB (dead after K gemm)
  _Float16* vals    = (_Float16*)(ws + 16 * MB);  // 8 MB (dead after V gemm)
  _Float16* Qh      = (_Float16*)(ws + 24 * MB);  // 8 MB
  _Float16* Kh      = (_Float16*)(ws + 32 * MB);  // 8 MB
  _Float16* Vth     = (_Float16*)(ws + 40 * MB);  // 8 MB
  _Float16* attnh   = (_Float16*)(ws + 0);        // reuses qpos
  float*    projf   = (float*)   (ws + 24 * MB);  // 16 MB, reuses Qh+Kh
  _Float16* attouth = (_Float16*)(ws + 8  * MB);  // reuses kpos
  float*    attoutf = (float*)   (ws + 48 * MB);  // 16 MB
  _Float16* ffh     = (_Float16*)(ws + 64 * MB);  // 32 MB
  float*    ff2f    = (float*)   (ws + 96 * MB);  // 16 MB
  _Float16* wqt     = (_Float16*)(ws + 112 * MB);
  _Float16* wkt = wqt + 512 * 512;
  _Float16* wvt = wkt + 512 * 512;
  _Float16* wot = wvt + 512 * 512;
  _Float16* w1t = wot + 512 * 512;
  _Float16* w2t = w1t + 512 * 2048;

  const int NE = ROWS * DM;
  k_addcvt<<<(NE + 255) / 256, 256, 0, stream>>>(queries, pos, qpos, NE);
  k_addcvt<<<(NE + 255) / 256, 256, 0, stream>>>(keys, pos, kpos, NE);
  k_addcvt<<<(NE + 255) / 256, 256, 0, stream>>>(values, nullptr, vals, NE);
  k_transcvt<<<(512 * 512 + 255) / 256, 256, 0, stream>>>(Wq, wqt, 512, 512);
  k_transcvt<<<(512 * 512 + 255) / 256, 256, 0, stream>>>(Wk, wkt, 512, 512);
  k_transcvt<<<(512 * 512 + 255) / 256, 256, 0, stream>>>(Wv, wvt, 512, 512);
  k_transcvt<<<(512 * 512 + 255) / 256, 256, 0, stream>>>(Wo, wot, 512, 512);
  k_transcvt<<<(512 * 2048 + 255) / 256, 256, 0, stream>>>(W1, w1t, 512, 2048);
  k_transcvt<<<(2048 * 512 + 255) / 256, 256, 0, stream>>>(W2, w2t, 2048, 512);

  dim3 blk(128);
  dim3 gqkv(DM / 64, ROWS / 64);
  k_gemm<<<gqkv, blk, 0, stream>>>(qpos, wqt, bq, Qh,  ROWS, DM, DM, GM_QK, 0);
  k_gemm<<<gqkv, blk, 0, stream>>>(kpos, wkt, bk, Kh,  ROWS, DM, DM, GM_QK, 0);
  k_gemm<<<gqkv, blk, 0, stream>>>(vals, wvt, bv, Vth, ROWS, DM, DM, GM_VT, 0);

  k_attn<<<BATCH * NH * (NTOK / 16), 32, 0, stream>>>(Qh, Kh, Vth, relw, attnh);

  k_gemm<<<gqkv, blk, 0, stream>>>(attnh, wot, bo, projf, ROWS, DM, DM, GM_F32, 0);
  k_ln<<<ROWS, 256, 0, stream>>>(projf, queries, g1, be1, attoutf, attouth);

  dim3 gff1(DFF / 64, ROWS / 64);
  k_gemm<<<gff1, blk, 0, stream>>>(attouth, w1t, b1, ffh, ROWS, DFF, DM, GM_F16, 1);
  dim3 gff2(DM / 64, ROWS / 64);
  k_gemm<<<gff2, blk, 0, stream>>>(ffh, w2t, b2, ff2f, ROWS, DM, DFF, GM_F32, 0);

  k_ln<<<ROWS, 256, 0, stream>>>(ff2f, attoutf, g2, be2, (float*)d_out, nullptr);
}
